// ResModel_46162308498229
// MI455X (gfx1250) — compile-verified, matrix operand order checked by reading, not run on hardware
//
#include <hip/hip_runtime.h>
#include <math.h>

// ---------------------------------------------------------------------------
// Model dims (from reference)
// ---------------------------------------------------------------------------
#define NLAYERS 10
#define DH      256
#define BB      64
#define LL      128
#define ROWS    (2 * BB * LL)   // 16384 token rows: [0,8192)=seq1, [8192,16384)=seq2
#define EMB     300
#define CCLS    3
static const float BN_SCALE_F = 0.99950037378f;  // 1/sqrt(1+1e-3)

typedef __bf16 v16bf __attribute__((ext_vector_type(16)));
typedef float  v8f   __attribute__((ext_vector_type(8)));

// Load a 16x32 bf16 A/B fragment row for v_wmma_f32_16x16x32_bf16.
// Per CDNA5 ISA layout: lanes 0-15 hold K={0..7, 16..23} (kb=0),
// lanes 16-31 hold K={8..15, 24..31} (kb=8). 'row' points at the 32 bf16
// K-values of this lane's matrix row in LDS, kb = ((lane>>4)<<3).
__device__ __forceinline__ v16bf frag16(const __bf16* row, int kb) {
  v16bf f;
#pragma unroll
  for (int i = 0; i < 8; ++i) {
    f[i]     = row[kb + i];
    f[i + 8] = row[kb + 16 + i];
  }
  return f;
}

__device__ __forceinline__ v8f v8f_zero() {
  v8f z = {0.f, 0.f, 0.f, 0.f, 0.f, 0.f, 0.f, 0.f};
  return z;
}

__device__ __forceinline__ float dev_sigmoid(float x) {
  return 1.0f / (1.0f + __expf(-x));
}

// ---------------------------------------------------------------------------
// Embedding gather: xemb[r, 0:300] = table[id(r)]
// ---------------------------------------------------------------------------
__global__ void embed_gather(const float* __restrict__ table,
                             const int* __restrict__ ids1,
                             const int* __restrict__ ids2,
                             float* __restrict__ xemb) {
  int r = blockIdx.x;
  int id = (r < BB * LL) ? ids1[r] : ids2[r - BB * LL];
  const float* src = table + (size_t)id * EMB;
  float* dst = xemb + (size_t)r * EMB;
  for (int c = threadIdx.x; c < EMB; c += blockDim.x) dst[c] = src[c];
}

// ---------------------------------------------------------------------------
// Row squared norms of h [ROWS, 256] -> sqn[ROWS]. One wave per row.
// ---------------------------------------------------------------------------
__global__ __launch_bounds__(256)
void row_sqnorm(const float* __restrict__ h, float* __restrict__ sqn) {
  int lane = threadIdx.x & 31;
  int wave = threadIdx.x >> 5;
  int row = blockIdx.x * 8 + wave;
  const float* r = h + (size_t)row * DH;
  float s = 0.f;
#pragma unroll
  for (int j = 0; j < 8; ++j) {
    float v = r[lane + 32 * j];
    s += v * v;
  }
#pragma unroll
  for (int off = 16; off > 0; off >>= 1) s += __shfl_xor(s, off, 32);
  if (lane == 0) sqn[row] = s;
}

// ---------------------------------------------------------------------------
// Generic WMMA GEMM with fused bias + inference-BN + activation.
//   C[M,N] = act( gmul * ((A @ W + bias) * bnscale) + badd )
// A is split into two K-segments (for the [h, attended] concat): A0 has K0
// columns (leading dim lda0), A1 has K1 columns. W is [K0+K1, N] row-major.
// act: 0=none, 1=relu, 2=sigmoid. gmul/badd may be null (=> 1 / 0).
// Block: 256 threads = 8 waves (2 M x 4 N), block tile 32M x 256N,
// wave tile 16M x 64N. grid = (M/32, N/256).
// ---------------------------------------------------------------------------
__global__ __launch_bounds__(256)
void gemm_bias_act(const float* __restrict__ A0, int lda0, int K0,
                   const float* __restrict__ A1, int lda1, int K1,
                   const float* __restrict__ W, int N,
                   const float* __restrict__ bias,
                   const float* __restrict__ gmul,
                   const float* __restrict__ badd,
                   float bnscale, int act,
                   float* __restrict__ C, int ldc, int M) {
  __shared__ __bf16 sA[32][40];    // [m][k], padded stride (16B aligned)
  __shared__ __bf16 sB[256][40];   // transposed weights: [n][k]

  const int tid   = threadIdx.x;
  const int lane  = tid & 31;
  const int wave  = tid >> 5;
  const int waveM = wave >> 2;   // 0..1
  const int waveN = wave & 3;    // 0..3
  const int blockM = blockIdx.x * 32;
  const int blockN = blockIdx.y * 256;
  const int K = K0 + K1;
  const int ktiles = (K + 31) >> 5;

  // staging decomposition (hoisted)
  const int am  = tid >> 3;          // 0..31  : A row
  const int ac0 = (tid & 7) << 2;    // 0..28  : A col quad
  const int bn0 = (tid & 63) << 2;   // 0..252 : W col quad
  const int bkq = tid >> 6;          // 0..3   : W row octet

  v8f acc[4];
#pragma unroll
  for (int i = 0; i < 4; ++i) acc[i] = v8f_zero();

  for (int kt = 0; kt < ktiles; ++kt) {
    const int k0 = kt << 5;
    // ---- stage A tile (32 rows x 32 K), f32 -> bf16
    {
      const float* Aseg;
      int lda, krel, segK;
      if (k0 < K0) { Aseg = A0; lda = lda0; krel = k0;      segK = K0; }
      else         { Aseg = A1; lda = lda1; krel = k0 - K0; segK = K1; }
      const float* src = Aseg + (size_t)(blockM + am) * lda + krel + ac0;
      if (krel + 32 <= segK) {  // full tile in range: vectorized, unguarded
        float4 v = *(const float4*)src;
        sA[am][ac0 + 0] = (__bf16)v.x;
        sA[am][ac0 + 1] = (__bf16)v.y;
        sA[am][ac0 + 2] = (__bf16)v.z;
        sA[am][ac0 + 3] = (__bf16)v.w;
      } else {                  // K tail (layer-0 K=300 only)
#pragma unroll
        for (int j = 0; j < 4; ++j) {
          int c = ac0 + j;
          float v = (krel + c < segK) ? src[j] : 0.0f;
          sA[am][c] = (__bf16)v;
        }
      }
    }
    // ---- stage W tile transposed: sB[n][k] = W[k0+k][blockN+n]
    //      8 float4 rows per thread, lane-consecutive n -> coalesced
    {
#pragma unroll
      for (int kk = 0; kk < 8; ++kk) {
        int kl = bkq * 8 + kk;
        int kg = k0 + kl;
        float4 v;
        if (kg < K) v = *(const float4*)(W + (size_t)kg * N + blockN + bn0);
        else        v = make_float4(0.f, 0.f, 0.f, 0.f);
        sB[bn0 + 0][kl] = (__bf16)v.x;
        sB[bn0 + 1][kl] = (__bf16)v.y;
        sB[bn0 + 2][kl] = (__bf16)v.z;
        sB[bn0 + 3][kl] = (__bf16)v.w;
      }
    }
    // prefetch next W tile into cache while this one is consumed
    if (kt + 1 < ktiles) {
      int kg = k0 + 32 + bkq * 8;
      if (kg < K) __builtin_prefetch(W + (size_t)kg * N + blockN + bn0, 0, 0);
    }
    __syncthreads();

    const int kb = (lane >> 4) << 3;
    v16bf af = frag16(&sA[waveM * 16 + (lane & 15)][0], kb);
#pragma unroll
    for (int nf = 0; nf < 4; ++nf) {
      v16bf bf = frag16(&sB[waveN * 64 + nf * 16 + (lane & 15)][0], kb);
      acc[nf] = __builtin_amdgcn_wmma_f32_16x16x32_bf16(
          false, af, false, bf, (short)0, acc[nf], false, false);
    }
    __syncthreads();
  }

  // ---- epilogue: C/D layout: VGPR r -> M=((lane>=16)?8:0)+r, N=lane&15
  const int mhalf = (lane >> 4) << 3;
#pragma unroll
  for (int nf = 0; nf < 4; ++nf) {
    int n = blockN + waveN * 64 + nf * 16 + (lane & 15);
    float bi = bias ? bias[n] : 0.0f;
    float gm = gmul ? gmul[n] : 1.0f;
    float ba = badd ? badd[n] : 0.0f;
#pragma unroll
    for (int r = 0; r < 8; ++r) {
      int m = blockM + waveM * 16 + mhalf + r;
      if (m < M) {
        float y = acc[nf][r] + bi;
        y = gm * (y * bnscale) + ba;
        if (act == 1)      y = fmaxf(y, 0.0f);
        else if (act == 2) y = dev_sigmoid(y);
        C[(size_t)m * ldc + n] = y;
      }
    }
  }
}

// ---------------------------------------------------------------------------
// Attention scores: per batch b, att[b] = 1/(1 + |h1_i|^2 + |h2_j|^2 - 2 h1·h2)
// One block per batch. 8 waves as 4M x 2N, wave tile 32M x 64N => 128x128.
// h2 row-major is exactly the transposed-B operand layout (n = h2 row).
// ---------------------------------------------------------------------------
__global__ __launch_bounds__(256)
void attn_scores(const float* __restrict__ h,
                 const float* __restrict__ sqn,
                 float* __restrict__ att) {
  __shared__ __bf16 s1[128][40];
  __shared__ __bf16 s2[128][40];
  __shared__ float lsq1[128];
  __shared__ float lsq2[128];

  const int b    = blockIdx.x;
  const int tid  = threadIdx.x;
  const int lane = tid & 31;
  const int wave = tid >> 5;
  const int waveM = wave >> 1;  // 0..3
  const int waveN = wave & 1;   // 0..1
  const float* h1 = h + (size_t)b * LL * DH;
  const float* h2 = h + (size_t)(BB * LL + b * LL) * DH;

  if (tid < 128) lsq1[tid] = sqn[b * LL + tid];
  else           lsq2[tid - 128] = sqn[BB * LL + b * LL + (tid - 128)];

  const int rq = tid >> 3;        // 0..31
  const int c0 = (tid & 7) << 2;  // 0..28

  v8f acc[2][4];
#pragma unroll
  for (int mf = 0; mf < 2; ++mf)
#pragma unroll
    for (int nf = 0; nf < 4; ++nf) acc[mf][nf] = v8f_zero();

  for (int kt = 0; kt < 8; ++kt) {
    const int k0 = kt << 5;
#pragma unroll
    for (int j = 0; j < 4; ++j) {
      int r = rq + (j << 5);  // 0..127
      float4 a = *(const float4*)(h1 + (size_t)r * DH + k0 + c0);
      float4 c = *(const float4*)(h2 + (size_t)r * DH + k0 + c0);
      s1[r][c0 + 0] = (__bf16)a.x;
      s1[r][c0 + 1] = (__bf16)a.y;
      s1[r][c0 + 2] = (__bf16)a.z;
      s1[r][c0 + 3] = (__bf16)a.w;
      s2[r][c0 + 0] = (__bf16)c.x;
      s2[r][c0 + 1] = (__bf16)c.y;
      s2[r][c0 + 2] = (__bf16)c.z;
      s2[r][c0 + 3] = (__bf16)c.w;
    }
    __syncthreads();

    const int kb = (lane >> 4) << 3;
    v16bf af0 = frag16(&s1[waveM * 32 + (lane & 15)][0], kb);
    v16bf af1 = frag16(&s1[waveM * 32 + 16 + (lane & 15)][0], kb);
#pragma unroll
    for (int nf = 0; nf < 4; ++nf) {
      v16bf bf = frag16(&s2[waveN * 64 + nf * 16 + (lane & 15)][0], kb);
      acc[0][nf] = __builtin_amdgcn_wmma_f32_16x16x32_bf16(
          false, af0, false, bf, (short)0, acc[0][nf], false, false);
      acc[1][nf] = __builtin_amdgcn_wmma_f32_16x16x32_bf16(
          false, af1, false, bf, (short)0, acc[1][nf], false, false);
    }
    __syncthreads();
  }

  const int mhalf = (lane >> 4) << 3;
  float* ab = att + (size_t)b * LL * LL;
#pragma unroll
  for (int mf = 0; mf < 2; ++mf) {
#pragma unroll
    for (int nf = 0; nf < 4; ++nf) {
      int n = waveN * 64 + nf * 16 + (lane & 15);
#pragma unroll
      for (int r = 0; r < 8; ++r) {
        int m = waveM * 32 + mf * 16 + mhalf + r;
        float d = lsq1[m] + lsq2[n] - 2.0f * acc[mf][nf][r];
        ab[(size_t)m * LL + n] = 1.0f / (1.0f + d);
      }
    }
  }
}

// ---------------------------------------------------------------------------
// Row softmax: P1[row, :] = softmax(att[row, :]), row = b*128 + i
// ---------------------------------------------------------------------------
__global__ __launch_bounds__(128)
void softmax_rows(const float* __restrict__ att, float* __restrict__ P) {
  __shared__ float red[128];
  int row = blockIdx.x;
  int t = threadIdx.x;
  float v = att[(size_t)row * LL + t];
  red[t] = v;
  __syncthreads();
  for (int s = 64; s > 0; s >>= 1) {
    if (t < s) red[t] = fmaxf(red[t], red[t + s]);
    __syncthreads();
  }
  float mx = red[0];
  __syncthreads();
  float e = __expf(v - mx);
  red[t] = e;
  __syncthreads();
  for (int s = 64; s > 0; s >>= 1) {
    if (t < s) red[t] += red[t + s];
    __syncthreads();
  }
  P[(size_t)row * LL + t] = e / red[0];
}

// ---------------------------------------------------------------------------
// Column softmax, written transposed: P2t[b, j, i] = softmax_i(att[b, i, j])
// ---------------------------------------------------------------------------
__global__ __launch_bounds__(128)
void softmax_cols_t(const float* __restrict__ att, float* __restrict__ Pt) {
  __shared__ float red[128];
  int b = blockIdx.x >> 7;
  int j = blockIdx.x & 127;
  int t = threadIdx.x;  // = i
  float v = att[((size_t)b * LL + t) * LL + j];
  red[t] = v;
  __syncthreads();
  for (int s = 64; s > 0; s >>= 1) {
    if (t < s) red[t] = fmaxf(red[t], red[t + s]);
    __syncthreads();
  }
  float mx = red[0];
  __syncthreads();
  float e = __expf(v - mx);
  red[t] = e;
  __syncthreads();
  for (int s = 64; s > 0; s >>= 1) {
    if (t < s) red[t] += red[t + s];
    __syncthreads();
  }
  Pt[((size_t)b * LL + j) * LL + t] = e / red[0];
}

// ---------------------------------------------------------------------------
// Attend: out[b, m, :] = P[b, m, :] @ H[b]   (P: [B,128,128], H: [B,128,256])
// grid = (B, 4): block tile 32M x 256N; 8 waves as 2M x 4N (16x64 per wave).
// ---------------------------------------------------------------------------
__global__ __launch_bounds__(256)
void attn_apply(const float* __restrict__ P,
                const float* __restrict__ H,
                float* __restrict__ out) {
  __shared__ __bf16 sP[32][40];
  __shared__ __bf16 sHt[256][40];  // [n][k]

  const int b  = blockIdx.x;
  const int mt = blockIdx.y;       // 0..3
  const int tid  = threadIdx.x;
  const int lane = tid & 31;
  const int wave = tid >> 5;
  const int waveM = wave >> 2;  // 0..1
  const int waveN = wave & 3;   // 0..3
  const float* Pb = P + (size_t)b * LL * LL + (size_t)mt * 32 * LL;
  const float* Hb = H + (size_t)b * LL * DH;
  float* Ob = out + ((size_t)b * LL + mt * 32) * DH;

  const int pr  = tid >> 3;          // 0..31
  const int pc0 = (tid & 7) << 2;    // 0..28
  const int hn0 = (tid & 63) << 2;   // 0..252
  const int hkq = tid >> 6;          // 0..3

  v8f acc[4];
#pragma unroll
  for (int i = 0; i < 4; ++i) acc[i] = v8f_zero();

  for (int kt = 0; kt < 4; ++kt) {
    const int k0 = kt << 5;
    {
      float4 v = *(const float4*)(Pb + (size_t)pr * LL + k0 + pc0);
      sP[pr][pc0 + 0] = (__bf16)v.x;
      sP[pr][pc0 + 1] = (__bf16)v.y;
      sP[pr][pc0 + 2] = (__bf16)v.z;
      sP[pr][pc0 + 3] = (__bf16)v.w;
    }
    {
#pragma unroll
      for (int kk = 0; kk < 8; ++kk) {
        int kl = hkq * 8 + kk;
        float4 v = *(const float4*)(Hb + (size_t)(k0 + kl) * DH + hn0);
        sHt[hn0 + 0][kl] = (__bf16)v.x;
        sHt[hn0 + 1][kl] = (__bf16)v.y;
        sHt[hn0 + 2][kl] = (__bf16)v.z;
        sHt[hn0 + 3][kl] = (__bf16)v.w;
      }
    }
    __syncthreads();

    const int kb = (lane >> 4) << 3;
    v16bf af = frag16(&sP[waveM * 16 + (lane & 15)][0], kb);
#pragma unroll
    for (int nf = 0; nf < 4; ++nf) {
      v16bf bf = frag16(&sHt[waveN * 64 + nf * 16 + (lane & 15)][0], kb);
      acc[nf] = __builtin_amdgcn_wmma_f32_16x16x32_bf16(
          false, af, false, bf, (short)0, acc[nf], false, false);
    }
    __syncthreads();
  }

  const int mhalf = (lane >> 4) << 3;
#pragma unroll
  for (int nf = 0; nf < 4; ++nf) {
    int n = waveN * 64 + nf * 16 + (lane & 15);
#pragma unroll
    for (int r = 0; r < 8; ++r) {
      int m = waveM * 16 + mhalf + r;
      Ob[(size_t)m * DH + n] = acc[nf][r];
    }
  }
}

// ---------------------------------------------------------------------------
// Gated residual combine: x = v*g + h*(1-g)   (vectorized float4)
// ---------------------------------------------------------------------------
__global__ void gate_combine(const float4* __restrict__ v,
                             const float4* __restrict__ g,
                             const float4* __restrict__ h,
                             float4* __restrict__ xo, int n4) {
  int i = blockIdx.x * blockDim.x + threadIdx.x;
  if (i < n4) {
    float4 vv = v[i], gg = g[i], hh = h[i];
    float4 o;
    o.x = vv.x * gg.x + hh.x * (1.0f - gg.x);
    o.y = vv.y * gg.y + hh.y * (1.0f - gg.y);
    o.z = vv.z * gg.z + hh.z * (1.0f - gg.z);
    o.w = vv.w * gg.w + hh.w * (1.0f - gg.w);
    xo[i] = o;
  }
}

// ---------------------------------------------------------------------------
// Pooling: feat[b] = [max_i x1, max_j x2, sum_i x1, sum_j x2]  -> [B, 1024]
// ---------------------------------------------------------------------------
__global__ __launch_bounds__(256)
void pool_feat(const float* __restrict__ x, float* __restrict__ feat) {
  int b = blockIdx.x;
  int d = threadIdx.x;  // 0..255
  const float* x1 = x + (size_t)b * LL * DH + d;
  const float* x2 = x + (size_t)(BB * LL + b * LL) * DH + d;
  float m1 = -3.4e38f, m2 = -3.4e38f, s1 = 0.f, s2 = 0.f;
  for (int i = 0; i < LL; ++i) {
    float v1 = x1[(size_t)i * DH];
    float v2 = x2[(size_t)i * DH];
    m1 = fmaxf(m1, v1);
    m2 = fmaxf(m2, v2);
    s1 += v1;
    s2 += v2;
  }
  float* f = feat + (size_t)b * 4 * DH;
  f[d] = m1;
  f[DH + d] = m2;
  f[2 * DH + d] = s1;
  f[3 * DH + d] = s2;
}

// ---------------------------------------------------------------------------
// Final projection: out[b, c] = t2[b] . Wout[:, c] + bout[c]   (64x1024x3)
// ---------------------------------------------------------------------------
__global__ void final_out(const float* __restrict__ t2,
                          const float* __restrict__ Wout,
                          const float* __restrict__ bout,
                          float* __restrict__ out) {
  int t = threadIdx.x;  // 0..191
  int b = t / CCLS, c = t % CCLS;
  float s = bout[c];
  const float* row = t2 + (size_t)b * 4 * DH;
  for (int k = 0; k < 4 * DH; ++k) s += row[k] * Wout[(size_t)k * CCLS + c];
  out[(size_t)b * CCLS + c] = s;
}

// ---------------------------------------------------------------------------
// Host launch
// ---------------------------------------------------------------------------
extern "C" void kernel_launch(void* const* d_in, const int* in_sizes, int n_in,
                              void* d_out, int out_size, void* d_ws, size_t ws_size,
                              hipStream_t stream) {
  (void)in_sizes; (void)n_in; (void)out_size; (void)ws_size;

  const float* embed    = (const float*)d_in[0];
  const float* W1_first = (const float*)d_in[1];
  const float* W1_rest  = (const float*)d_in[2];
  const float* b1    = (const float*)d_in[3];
  const float* bn1_g = (const float*)d_in[4];
  const float* bn1_b = (const float*)d_in[5];
  const float* W2    = (const float*)d_in[6];
  const float* b2    = (const float*)d_in[7];
  const float* bn2_g = (const float*)d_in[8];
  const float* bn2_b = (const float*)d_in[9];
  const float* Wg    = (const float*)d_in[10];
  const float* bg    = (const float*)d_in[11];
  const float* Wf1   = (const float*)d_in[12];
  const float* bf1   = (const float*)d_in[13];
  const float* Wf2   = (const float*)d_in[14];
  const float* bf2   = (const float*)d_in[15];
  const float* Wout  = (const float*)d_in[16];
  const float* bout  = (const float*)d_in[17];
  const int* x1_ids  = (const int*)d_in[18];
  const int* x2_ids  = (const int*)d_in[19];

  float* ws = (float*)d_ws;
  size_t o = 0;
  float* xemb     = ws + o; o += (size_t)ROWS * EMB;
  float* xcur     = ws + o; o += (size_t)ROWS * DH;
  float* hbuf     = ws + o; o += (size_t)ROWS * DH;
  float* attended = ws + o; o += (size_t)ROWS * DH;
  float* vbuf     = ws + o; o += (size_t)ROWS * DH;
  float* gbuf     = ws + o; o += (size_t)ROWS * DH;
  float* attw     = ws + o; o += (size_t)BB * LL * LL;
  float* P1       = ws + o; o += (size_t)BB * LL * LL;
  float* P2t      = ws + o; o += (size_t)BB * LL * LL;
  float* sqn      = ws + o; o += (size_t)ROWS;
  float* feat     = ws + o; o += (size_t)BB * 4 * DH;
  float* t1       = ws + o; o += (size_t)BB * 4 * DH;
  float* t2       = ws + o; o += (size_t)BB * 4 * DH;

  embed_gather<<<ROWS, 128, 0, stream>>>(embed, x1_ids, x2_ids, xemb);

  for (int i = 0; i < NLAYERS; ++i) {
    const float* W1 = (i == 0) ? W1_first : (W1_rest + (size_t)(i - 1) * DH * DH);
    const float* A  = (i == 0) ? xemb : xcur;
    const int K     = (i == 0) ? EMB : DH;

    // h = relu(BN(x @ W1 + b1))   [16384, 256]
    gemm_bias_act<<<dim3(ROWS / 32, 1), 256, 0, stream>>>(
        A, K, K, nullptr, 0, 0, W1, DH,
        b1 + i * DH, bn1_g + i * DH, bn1_b + i * DH, BN_SCALE_F, 1,
        hbuf, DH, ROWS);

    row_sqnorm<<<ROWS / 8, 256, 0, stream>>>(hbuf, sqn);
    attn_scores<<<BB, 256, 0, stream>>>(hbuf, sqn, attw);
    softmax_rows<<<BB * LL, 128, 0, stream>>>(attw, P1);
    softmax_cols_t<<<BB * LL, 128, 0, stream>>>(attw, P2t);

    // beta  = P1  @ h2  -> attended rows [0, 8192)
    attn_apply<<<dim3(BB, 4), 256, 0, stream>>>(
        P1, hbuf + (size_t)BB * LL * DH, attended);
    // alpha = P2t @ h1  -> attended rows [8192, 16384)
    attn_apply<<<dim3(BB, 4), 256, 0, stream>>>(
        P2t, hbuf, attended + (size_t)BB * LL * DH);

    // v = relu(BN([h, attended] @ W2 + b2))
    gemm_bias_act<<<dim3(ROWS / 32, 1), 256, 0, stream>>>(
        hbuf, DH, DH, attended, DH, DH, W2 + (size_t)i * 2 * DH * DH, DH,
        b2 + i * DH, bn2_g + i * DH, bn2_b + i * DH, BN_SCALE_F, 1,
        vbuf, DH, ROWS);

    // g = sigmoid(h @ Wg + bg)
    gemm_bias_act<<<dim3(ROWS / 32, 1), 256, 0, stream>>>(
        hbuf, DH, DH, nullptr, 0, 0, Wg + (size_t)i * DH * DH, DH,
        bg + i * DH, nullptr, nullptr, 1.0f, 2,
        gbuf, DH, ROWS);

    // x = v*g + h*(1-g)
    int n4 = ROWS * DH / 4;
    gate_combine<<<(n4 + 255) / 256, 256, 0, stream>>>(
        (const float4*)vbuf, (const float4*)gbuf, (const float4*)hbuf,
        (float4*)xcur, n4);
  }

  pool_feat<<<BB, 256, 0, stream>>>(xcur, feat);

  gemm_bias_act<<<dim3(BB / 32, 4), 256, 0, stream>>>(
      feat, 4 * DH, 4 * DH, nullptr, 0, 0, Wf1, 4 * DH,
      bf1, nullptr, nullptr, 1.0f, 1, t1, 4 * DH, BB);
  gemm_bias_act<<<dim3(BB / 32, 4), 256, 0, stream>>>(
      t1, 4 * DH, 4 * DH, nullptr, 0, 0, Wf2, 4 * DH,
      bf2, nullptr, nullptr, 1.0f, 1, t2, 4 * DH, BB);

  final_out<<<1, BB * CCLS, 0, stream>>>(t2, Wout, bout, (float*)d_out);
}
